// TransformerEncoderLayer_7533372637372
// MI455X (gfx1250) — compile-verified, matrix operand order checked by reading, not run on hardware
//
#include <hip/hip_runtime.h>
#include <hip/hip_bf16.h>

typedef __attribute__((ext_vector_type(16))) _Float16 v16h;
typedef __attribute__((ext_vector_type(8)))  _Float16 h8;
typedef __attribute__((ext_vector_type(8)))  float    v8f;

#define TOPK 1000
#define DMODEL 256

static __device__ __forceinline__ v16h concat8(h8 a, h8 b) {
  return __builtin_shufflevector(a, b, 0, 1, 2, 3, 4, 5, 6, 7, 8, 9, 10, 11, 12, 13, 14, 15);
}
static __device__ __forceinline__ h8 cvt8(float4 a, float4 b) {
  h8 r;
  r[0] = (_Float16)a.x; r[1] = (_Float16)a.y; r[2] = (_Float16)a.z; r[3] = (_Float16)a.w;
  r[4] = (_Float16)b.x; r[5] = (_Float16)b.y; r[6] = (_Float16)b.z; r[7] = (_Float16)b.w;
  return r;
}

// ---------------------------------------------------------------------------
// Per-batch top-k (k=1000) via 256-bin float-bit histogram (scores in (0,1)).
// ---------------------------------------------------------------------------
__global__ __launch_bounds__(256) void topk_kernel(
    const float* __restrict__ sal, const float* __restrict__ prob,
    const long long* __restrict__ boffs, int* __restrict__ selidx)
{
  int b = blockIdx.x;
  int start = (int)boffs[b], end = (int)boffs[b + 1];
  __shared__ int hist[256];
  __shared__ int s_thb, s_above, s_hi, s_tie;
  for (int i = threadIdx.x; i < 256; i += 256) hist[i] = 0;
  if (threadIdx.x == 0) { s_hi = 0; s_tie = 0; }
  __syncthreads();
  for (int t = start + threadIdx.x; t < end; t += 256) {
    float s = prob[t] * (1.f / (1.f + __expf(-sal[t])));
    unsigned u = __float_as_uint(s);
    atomicAdd(&hist[u >> 24], 1);
  }
  __syncthreads();
  if (threadIdx.x == 0) {
    int cum = 0, thb = 0, above = 0;
    for (int bkt = 255; bkt >= 0; --bkt) {
      int c = hist[bkt];
      if (cum + c >= TOPK || bkt == 0) { thb = bkt; above = cum; break; }
      cum += c;
    }
    s_thb = thb; s_above = above;
  }
  for (int i = threadIdx.x; i < TOPK; i += 256) selidx[b * TOPK + i] = -1;
  __syncthreads();
  int thb = s_thb, above = s_above;
  for (int t = start + threadIdx.x; t < end; t += 256) {
    float s = prob[t] * (1.f / (1.f + __expf(-sal[t])));
    int bkt = (int)(__float_as_uint(s) >> 24);
    if (bkt > thb) {
      int pos = atomicAdd(&s_hi, 1);
      if (pos < TOPK) selidx[b * TOPK + pos] = t;
    } else if (bkt == thb) {
      int pos = above + atomicAdd(&s_tie, 1);
      if (pos < TOPK) selidx[b * TOPK + pos] = t;
    }
  }
}

// ---------------------------------------------------------------------------
// Gather selected rows of queries / pos-encodings.
// ---------------------------------------------------------------------------
__global__ void gather_kernel(const float* __restrict__ q, const float* __restrict__ p,
                              const int* __restrict__ selidx,
                              float* __restrict__ selQ, float* __restrict__ selP, int nSel)
{
  int i = blockIdx.x * 256 + threadIdx.x;
  if (i >= nSel * DMODEL) return;
  int s = i >> 8, c = i & 255;
  int t = selidx[s];
  selQ[i] = (t >= 0) ? q[(size_t)t * DMODEL + c] : 0.f;
  selP[i] = (t >= 0) ? p[(size_t)t * DMODEL + c] : 0.f;
}

// ---------------------------------------------------------------------------
// LayerNorm over rows of 256, one wave per row (wave32; 8 elems/lane).
// ---------------------------------------------------------------------------
__global__ __launch_bounds__(256) void ln_kernel(
    const float* __restrict__ X, const float* __restrict__ P,
    const float* __restrict__ g, const float* __restrict__ bb,
    float* __restrict__ Y, int rows)
{
  int row = blockIdx.x * 8 + (threadIdx.x >> 5);
  int lane = threadIdx.x & 31;
  if (row >= rows) return;
  const float* x = X + (size_t)row * DMODEL;
  float v[8], s = 0.f, sq = 0.f;
  #pragma unroll
  for (int j = 0; j < 8; ++j) { float t = x[lane + 32 * j]; v[j] = t; s += t; sq += t * t; }
  #pragma unroll
  for (int m = 16; m; m >>= 1) { s += __shfl_xor(s, m, 32); sq += __shfl_xor(sq, m, 32); }
  float mean = s * (1.f / 256.f);
  float var = sq * (1.f / 256.f) - mean * mean;
  float inv = rsqrtf(var + 1e-5f);
  float* y = Y + (size_t)row * DMODEL;
  #pragma unroll
  for (int j = 0; j < 8; ++j) {
    int c = lane + 32 * j;
    float r = (v[j] - mean) * inv * g[c] + bb[c];
    if (P) r += P[(size_t)row * DMODEL + c];
    y[c] = r;
  }
}

// ---------------------------------------------------------------------------
// Tiled GEMM: C[M,N] = act(A[M,K] @ B[K,N] + bias) + resid.
// fp32 in/out, f16 WMMA compute. REQUIRES: K % 32 == 0, N % 64 == 0,
// lda/ldb/ldc % 4 == 0 (true for all launches below).
// Block = 256 threads = 8 waves; block tile 64x64; wave tile 16x32 (2 WMMAs).
// A staged row-major, B staged transposed (n,k): all fragments are b128 reads.
// ---------------------------------------------------------------------------
__global__ __launch_bounds__(256) void gemm_wmma(
    const float* __restrict__ A, int lda,
    const float* __restrict__ Bm, int ldb,
    float* __restrict__ C, int ldc,
    int M, int N, int K,
    const float* __restrict__ bias,
    const float* __restrict__ resid,
    int act)
{
  __shared__ __align__(16) _Float16 As[64][40];
  __shared__ __align__(16) _Float16 Bt[64][40];   // (n, k) transposed
  int tid = threadIdx.x;
  int wave = tid >> 5, lane = tid & 31;
  int m0 = blockIdx.y * 64, n0 = blockIdx.x * 64;
  int mOff = (wave & 3) * 16;
  int nOff = (wave >> 2) * 32;
  int laneIdx = lane & 15, laneHalf = lane >> 4;
  v8f acc0 = {}, acc1 = {};
  int arow = tid >> 2, acol = (tid & 3) * 8;      // A: 64x32, 8 elems/thread
  int bn = tid & 63, bk = (tid >> 6) * 8;         // B: 32x64, 8 elems/thread

  for (int k0 = 0; k0 < K; k0 += 32) {
    {   // A tile: two global b128 loads under a single per-row predicate
      int gm = m0 + arow;
      float4 u0 = {}, u1 = {};
      if (gm < M) {
        const float4* s4 = reinterpret_cast<const float4*>(A + (size_t)gm * lda + k0 + acol);
        u0 = s4[0]; u1 = s4[1];
        if (k0 + 32 < K) __builtin_prefetch(s4 + 8, 0, 1);   // global_prefetch_b8
      }
      *reinterpret_cast<h8*>(&As[arow][acol]) = cvt8(u0, u1);
    }
    {   // B tile: column-wise coalesced loads, stored transposed, one b128 ds store
      const float* bp = Bm + (size_t)(k0 + bk) * ldb + n0 + bn;
      h8 hv;
      #pragma unroll
      for (int j = 0; j < 8; ++j) hv[j] = (_Float16)bp[(size_t)j * ldb];
      *reinterpret_cast<h8*>(&Bt[bn][bk]) = hv;
    }
    __syncthreads();
    h8 a0 = *reinterpret_cast<const h8*>(&As[mOff + laneIdx][laneHalf * 8]);
    h8 a1 = *reinterpret_cast<const h8*>(&As[mOff + laneIdx][16 + laneHalf * 8]);
    v16h af = concat8(a0, a1);
    h8 b0 = *reinterpret_cast<const h8*>(&Bt[nOff + laneIdx][laneHalf * 8]);
    h8 b1 = *reinterpret_cast<const h8*>(&Bt[nOff + laneIdx][16 + laneHalf * 8]);
    h8 b2 = *reinterpret_cast<const h8*>(&Bt[nOff + 16 + laneIdx][laneHalf * 8]);
    h8 b3 = *reinterpret_cast<const h8*>(&Bt[nOff + 16 + laneIdx][16 + laneHalf * 8]);
    acc0 = __builtin_amdgcn_wmma_f32_16x16x32_f16(false, af, false, concat8(b0, b1),
                                                  (short)0, acc0, false, false);
    acc1 = __builtin_amdgcn_wmma_f32_16x16x32_f16(false, af, false, concat8(b2, b3),
                                                  (short)0, acc1, false, false);
    __syncthreads();
  }

  int gn0 = n0 + nOff + laneIdx;
  int gn1 = gn0 + 16;
  float bv0 = bias ? bias[gn0] : 0.f;
  float bv1 = bias ? bias[gn1] : 0.f;
  #pragma unroll
  for (int r = 0; r < 8; ++r) {
    int gm = m0 + mOff + laneHalf * 8 + r;
    if (gm < M) {
      float v0 = acc0[r] + bv0;
      float v1 = acc1[r] + bv1;
      if (act == 1) {
        v0 = 0.5f * v0 * (1.f + erff(v0 * 0.70710678118f));
        v1 = 0.5f * v1 * (1.f + erff(v1 * 0.70710678118f));
      }
      if (resid) {
        v0 += resid[(size_t)gm * ldc + gn0];
        v1 += resid[(size_t)gm * ldc + gn1];
      }
      C[(size_t)gm * ldc + gn0] = v0;
      C[(size_t)gm * ldc + gn1] = v1;
    }
  }
}

// ---------------------------------------------------------------------------
// Flash-attention over selected tokens. 1 wave per (batch, head, 16-row q-tile).
// QK: [B*nQ, 512] (q cols 0..255, k cols 256..511); V: [B*nQ, 256]; O: [B*nQ, 256].
// Q/K fragments from contiguous float4 global loads; V staged through LDS.
// ---------------------------------------------------------------------------
__global__ __launch_bounds__(32) void attn_kernel(
    const float* __restrict__ QK, const float* __restrict__ V,
    float* __restrict__ O, int nQ)
{
  __shared__ __align__(16) _Float16 Pt[16][40];
  __shared__ __align__(16) _Float16 Vs[32][40];
  int lane = threadIdx.x;
  int laneIdx = lane & 15, laneHalf = lane >> 4;
  int nQT = (nQ + 15) / 16;
  int qt = blockIdx.x % nQT;
  int h  = (blockIdx.x / nQT) & 7;
  int b  = blockIdx.x / (nQT * 8);
  const float scale = 0.17677669529663687f;  // 1/sqrt(32)

  int qrow = qt * 16 + laneIdx;
  bool qok = qrow < nQ;
  v16h qf;
  {
    float4 a = {}, bq = {}, c = {}, d = {};
    if (qok) {
      const float* qbase = QK + ((size_t)(b * nQ + qrow)) * 512 + h * 32;
      const float4* p0 = reinterpret_cast<const float4*>(qbase + laneHalf * 8);
      const float4* p1 = reinterpret_cast<const float4*>(qbase + 16 + laneHalf * 8);
      a = p0[0]; bq = p0[1]; c = p1[0]; d = p1[1];
    }
    a.x *= scale; a.y *= scale; a.z *= scale; a.w *= scale;
    bq.x *= scale; bq.y *= scale; bq.z *= scale; bq.w *= scale;
    c.x *= scale; c.y *= scale; c.z *= scale; c.w *= scale;
    d.x *= scale; d.y *= scale; d.z *= scale; d.w *= scale;
    qf = concat8(cvt8(a, bq), cvt8(c, d));
  }
  float mrow[8], lrow[8];
  v8f o0 = {}, o1 = {};
  #pragma unroll
  for (int r = 0; r < 8; ++r) { mrow[r] = -1e30f; lrow[r] = 0.f; }

  for (int kt = 0; kt < nQ; kt += 32) {
    // K fragments (B-layout: lane = key column, elems walk head-dim)
    int kk0 = kt + laneIdx, kk1 = kt + 16 + laneIdx;
    bool v0 = kk0 < nQ, v1 = kk1 < nQ;
    v16h kf0, kf1;
    {
      float4 a = {}, bq = {}, c = {}, d = {};
      if (v0) {
        const float* kb = QK + ((size_t)(b * nQ + kk0)) * 512 + 256 + h * 32;
        const float4* p0 = reinterpret_cast<const float4*>(kb + laneHalf * 8);
        const float4* p1 = reinterpret_cast<const float4*>(kb + 16 + laneHalf * 8);
        a = p0[0]; bq = p0[1]; c = p1[0]; d = p1[1];
      }
      kf0 = concat8(cvt8(a, bq), cvt8(c, d));
    }
    {
      float4 a = {}, bq = {}, c = {}, d = {};
      if (v1) {
        const float* kb = QK + ((size_t)(b * nQ + kk1)) * 512 + 256 + h * 32;
        const float4* p0 = reinterpret_cast<const float4*>(kb + laneHalf * 8);
        const float4* p1 = reinterpret_cast<const float4*>(kb + 16 + laneHalf * 8);
        a = p0[0]; bq = p0[1]; c = p1[0]; d = p1[1];
      }
      kf1 = concat8(cvt8(a, bq), cvt8(c, d));
    }
    // Stage V tile (32 keys x 32 dims): one row per lane, b128 stores
    {
      int kk = kt + lane;
      float4 u[8] = {};
      if (kk < nQ) {
        const float4* vp = reinterpret_cast<const float4*>(V + ((size_t)(b * nQ + kk)) * 256 + h * 32);
        #pragma unroll
        for (int j = 0; j < 8; ++j) u[j] = vp[j];
      }
      #pragma unroll
      for (int j = 0; j < 4; ++j)
        *reinterpret_cast<h8*>(&Vs[lane][j * 8]) = cvt8(u[2 * j], u[2 * j + 1]);
    }

    v8f z = {};
    v8f s0 = __builtin_amdgcn_wmma_f32_16x16x32_f16(false, qf, false, kf0, (short)0, z, false, false);
    v8f s1 = __builtin_amdgcn_wmma_f32_16x16x32_f16(false, qf, false, kf1, (short)0, z, false, false);
    #pragma unroll
    for (int r = 0; r < 8; ++r) {
      if (!v0) s0[r] = -1e9f;
      if (!v1) s1[r] = -1e9f;
    }
    // online softmax: each row of a C-tile lives across a 16-lane half-wave
    #pragma unroll
    for (int r = 0; r < 8; ++r) {
      float lm = fmaxf(s0[r], s1[r]);
      #pragma unroll
      for (int mm = 8; mm; mm >>= 1) lm = fmaxf(lm, __shfl_xor(lm, mm, 32));
      float mn = fmaxf(mrow[r], lm);
      float alpha = __expf(mrow[r] - mn);
      float p0 = __expf(s0[r] - mn);
      float p1 = __expf(s1[r] - mn);
      float psum = p0 + p1;
      #pragma unroll
      for (int mm = 8; mm; mm >>= 1) psum += __shfl_xor(psum, mm, 32);
      lrow[r] = lrow[r] * alpha + psum;
      mrow[r] = mn;
      o0[r] *= alpha; o1[r] *= alpha;
      Pt[laneHalf * 8 + r][laneIdx] = (_Float16)p0;
      Pt[laneHalf * 8 + r][16 + laneIdx] = (_Float16)p1;
    }
    __syncthreads();
    h8 pa = *reinterpret_cast<const h8*>(&Pt[laneIdx][laneHalf * 8]);
    h8 pb = *reinterpret_cast<const h8*>(&Pt[laneIdx][16 + laneHalf * 8]);
    v16h pf = concat8(pa, pb);
    v16h vf0, vf1;
    #pragma unroll
    for (int i = 0; i < 8; ++i) {
      vf0[i]     = Vs[laneHalf * 8 + i][laneIdx];
      vf0[i + 8] = Vs[16 + laneHalf * 8 + i][laneIdx];
      vf1[i]     = Vs[laneHalf * 8 + i][16 + laneIdx];
      vf1[i + 8] = Vs[16 + laneHalf * 8 + i][16 + laneIdx];
    }
    o0 = __builtin_amdgcn_wmma_f32_16x16x32_f16(false, pf, false, vf0, (short)0, o0, false, false);
    o1 = __builtin_amdgcn_wmma_f32_16x16x32_f16(false, pf, false, vf1, (short)0, o1, false, false);
    __syncthreads();
  }
  #pragma unroll
  for (int r = 0; r < 8; ++r) {
    int m = qt * 16 + laneHalf * 8 + r;
    if (m < nQ) {
      float inv = lrow[r] > 0.f ? 1.f / lrow[r] : 0.f;
      float* op = O + ((size_t)(b * nQ + m)) * 256 + h * 32;
      op[laneIdx] = o0[r] * inv;
      op[16 + laneIdx] = o1[r] * inv;
    }
  }
}

// ---------------------------------------------------------------------------
// Scatter SA result (residual) back into q2.
// ---------------------------------------------------------------------------
__global__ void scatter_kernel(const int* __restrict__ selidx,
                               const float* __restrict__ selQ,
                               const float* __restrict__ sa,
                               float* __restrict__ q2, int nSel)
{
  int i = blockIdx.x * 256 + threadIdx.x;
  if (i >= nSel * DMODEL) return;
  int s = i >> 8, c = i & 255;
  int t = selidx[s];
  if (t >= 0) q2[(size_t)t * DMODEL + c] = selQ[i] + sa[i];
}

// ---------------------------------------------------------------------------
// Softmax over 16 (level,point) logits per (token, head).
// ---------------------------------------------------------------------------
__global__ void softmax16_kernel(float* __restrict__ aw, int groups)
{
  int g = blockIdx.x * blockDim.x + threadIdx.x;
  if (g >= groups) return;
  float* p = aw + (size_t)g * 16;
  float m = -1e30f;
  #pragma unroll
  for (int i = 0; i < 16; ++i) m = fmaxf(m, p[i]);
  float e[16], s = 0.f;
  #pragma unroll
  for (int i = 0; i < 16; ++i) { e[i] = __expf(p[i] - m); s += e[i]; }
  float inv = 1.f / s;
  #pragma unroll
  for (int i = 0; i < 16; ++i) p[i] = e[i] * inv;
}

// ---------------------------------------------------------------------------
// Multi-scale deformable sampling. One wave per (token, head); lane = channel.
// ---------------------------------------------------------------------------
__global__ __launch_bounds__(256) void deform_kernel(
    const float* __restrict__ val, const float* __restrict__ offb,
    const float* __restrict__ awb, const float* __restrict__ xy,
    const long long* __restrict__ boffs, float* __restrict__ dout, int T)
{
  int wid = blockIdx.x * 8 + (threadIdx.x >> 5);
  int lane = threadIdx.x & 31;
  int t = wid >> 3, h = wid & 7;
  if (t >= T) return;
  int bid = 0;
  long long tt = t;
  if (tt >= boffs[1]) bid = 1;
  if (tt >= boffs[2]) bid = 2;
  if (tt >= boffs[3]) bid = 3;
  const int HWs[4] = {256, 128, 64, 32};
  const int LS[4]  = {0, 65536, 81920, 86016};
  float cx = xy[t * 2 + 0], cy = xy[t * 2 + 1];
  float acc = 0.f;
  #pragma unroll
  for (int l = 0; l < 4; ++l) {
    int W = HWs[l], H = HWs[l];
    const float* base = val + ((size_t)bid * 87040 + LS[l]) * 256 + h * 32 + lane;
    #pragma unroll
    for (int p = 0; p < 4; ++p) {
      int oi = t * 256 + h * 32 + l * 8 + p * 2;
      float ox = offb[oi], oy = offb[oi + 1];
      float a = awb[t * 128 + h * 16 + l * 4 + p];
      float x = (cx + ox / (float)W) * W - 0.5f;
      float y = (cy + oy / (float)H) * H - 0.5f;
      float x0f = floorf(x), y0f = floorf(y);
      float wx = x - x0f, wy = y - y0f;
      int x0 = (int)x0f, y0 = (int)y0f;
      float s = 0.f;
      #pragma unroll
      for (int cyi = 0; cyi < 2; ++cyi)
        #pragma unroll
        for (int cxi = 0; cxi < 2; ++cxi) {
          int xi = x0 + cxi, yi = y0 + cyi;
          if (xi < 0 || xi >= W || yi < 0 || yi >= H) continue;
          float wgt = (cxi ? wx : 1.f - wx) * (cyi ? wy : 1.f - wy);
          s += wgt * base[((size_t)yi * W + xi) * 256];
        }
      acc += a * s;
    }
  }
  dout[(size_t)t * 256 + h * 32 + lane] = acc;
}

// ---------------------------------------------------------------------------
// Host-side orchestration.
// ---------------------------------------------------------------------------
extern "C" void kernel_launch(void* const* d_in, const int* in_sizes, int n_in,
                              void* d_out, int out_size, void* d_ws, size_t ws_size,
                              hipStream_t stream) {
  (void)n_in; (void)out_size; (void)ws_size;
  const float* queries  = (const float*)d_in[0];
  const float* qpos     = (const float*)d_in[1];
  const float* xy       = (const float*)d_in[3];
  const long long* boff = (const long long*)d_in[4];
  const float* fmaps    = (const float*)d_in[5];
  const float* sal      = (const float*)d_in[7];
  const float* prob     = (const float*)d_in[8];
  const float* sa_ln_g  = (const float*)d_in[10];
  const float* sa_ln_b  = (const float*)d_in[11];
  const float* sa_qkv_w = (const float*)d_in[12];
  const float* sa_out_w = (const float*)d_in[13];
  const float* d_ln_g   = (const float*)d_in[14];
  const float* d_ln_b   = (const float*)d_in[15];
  const float* d_val_w  = (const float*)d_in[16];
  const float* d_val_b  = (const float*)d_in[17];
  const float* d_off_w  = (const float*)d_in[18];
  const float* d_off_b  = (const float*)d_in[19];
  const float* d_attn_w = (const float*)d_in[20];
  const float* d_attn_b = (const float*)d_in[21];
  const float* d_out_w  = (const float*)d_in[22];
  const float* d_out_b  = (const float*)d_in[23];
  const float* f_ln_g   = (const float*)d_in[24];
  const float* f_ln_b   = (const float*)d_in[25];
  const float* ffn_w1   = (const float*)d_in[26];
  const float* ffn_b1   = (const float*)d_in[27];
  const float* ffn_w2   = (const float*)d_in[28];
  const float* ffn_b2   = (const float*)d_in[29];

  int T = in_sizes[0] / DMODEL;                 // 50000
  int B = in_sizes[4] - 1;                      // 4
  int HWtot = in_sizes[5] / (B * DMODEL);       // 87040
  int nSel = B * TOPK;                          // 4000

  size_t off = 0;
  auto carve = [&](size_t elems, size_t esz) -> void* {
    void* p = (char*)d_ws + off;
    off += (elems * esz + 255) & ~(size_t)255;
    return p;
  };
  int*   selidx = (int*)carve((size_t)nSel, 4);
  float* selQ  = (float*)carve((size_t)nSel * 256, 4);
  float* selP  = (float*)carve((size_t)nSel * 256, 4);
  float* selX  = (float*)carve((size_t)nSel * 256, 4);
  float* selXP = (float*)carve((size_t)nSel * 256, 4);
  float* qk    = (float*)carve((size_t)nSel * 512, 4);
  float* vv    = (float*)carve((size_t)nSel * 256, 4);
  float* attnO = (float*)carve((size_t)nSel * 256, 4);
  float* sa    = (float*)carve((size_t)nSel * 256, 4);
  float* q2    = (float*)carve((size_t)T * 256, 4);
  float* xp    = (float*)carve((size_t)T * 256, 4);
  float* valb  = (float*)carve((size_t)B * HWtot * 256, 4);
  float* offb  = (float*)carve((size_t)T * 256, 4);
  float* awb   = (float*)carve((size_t)T * 128, 4);
  float* doutb = (float*)carve((size_t)T * 256, 4);
  float* q3    = (float*)carve((size_t)T * 256, 4);
  float* x3    = (float*)carve((size_t)T * 256, 4);
  float* hid   = (float*)carve((size_t)T * 1024, 4);

  auto gemm_grid = [](int M, int N) {
    return dim3((unsigned)((N + 63) / 64), (unsigned)((M + 63) / 64));
  };

  // ---- salience top-k + gather + self-attention over selected tokens ----
  topk_kernel<<<B, 256, 0, stream>>>(sal, prob, boff, selidx);
  gather_kernel<<<(nSel * 256 + 255) / 256, 256, 0, stream>>>(queries, qpos, selidx, selQ, selP, nSel);
  ln_kernel<<<(nSel + 7) / 8, 256, 0, stream>>>(selQ, nullptr, sa_ln_g, sa_ln_b, selX, nSel);
  ln_kernel<<<(nSel + 7) / 8, 256, 0, stream>>>(selQ, selP, sa_ln_g, sa_ln_b, selXP, nSel);
  // QK = (x+p) @ [Wq|Wk]  (cols 0..511 of sa_qkv_w)
  gemm_wmma<<<gemm_grid(nSel, 512), 256, 0, stream>>>(selXP, 256, sa_qkv_w, 768, qk, 512,
                                                      nSel, 512, 256, nullptr, nullptr, 0);
  // V = x @ Wv  (cols 512..767)
  gemm_wmma<<<gemm_grid(nSel, 256), 256, 0, stream>>>(selX, 256, sa_qkv_w + 512, 768, vv, 256,
                                                      nSel, 256, 256, nullptr, nullptr, 0);
  {
    int nQT = (TOPK + 15) / 16;
    attn_kernel<<<B * 8 * nQT, 32, 0, stream>>>(qk, vv, attnO, TOPK);
  }
  gemm_wmma<<<gemm_grid(nSel, 256), 256, 0, stream>>>(attnO, 256, sa_out_w, 256, sa, 256,
                                                      nSel, 256, 256, nullptr, nullptr, 0);
  hipMemcpyAsync(q2, queries, (size_t)T * 256 * sizeof(float), hipMemcpyDeviceToDevice, stream);
  scatter_kernel<<<(nSel * 256 + 255) / 256, 256, 0, stream>>>(selidx, selQ, sa, q2, nSel);

  // ---- deformable cross-attention ----
  ln_kernel<<<(T + 7) / 8, 256, 0, stream>>>(q2, qpos, d_ln_g, d_ln_b, xp, T);
  gemm_wmma<<<gemm_grid(B * HWtot, 256), 256, 0, stream>>>(fmaps, 256, d_val_w, 256, valb, 256,
                                                           B * HWtot, 256, 256, d_val_b, nullptr, 0);
  gemm_wmma<<<gemm_grid(T, 256), 256, 0, stream>>>(xp, 256, d_off_w, 256, offb, 256,
                                                   T, 256, 256, d_off_b, nullptr, 0);
  gemm_wmma<<<gemm_grid(T, 128), 256, 0, stream>>>(xp, 256, d_attn_w, 128, awb, 128,
                                                   T, 128, 256, d_attn_b, nullptr, 0);
  softmax16_kernel<<<(T * 8 + 255) / 256, 256, 0, stream>>>(awb, T * 8);
  deform_kernel<<<T, 256, 0, stream>>>(valb, offb, awb, xy, boff, doutb, T);
  gemm_wmma<<<gemm_grid(T, 256), 256, 0, stream>>>(doutb, 256, d_out_w, 256, q3, 256,
                                                   T, 256, 256, d_out_b, q2, 0);

  // ---- FFN ----
  ln_kernel<<<(T + 7) / 8, 256, 0, stream>>>(q3, nullptr, f_ln_g, f_ln_b, x3, T);
  gemm_wmma<<<gemm_grid(T, 1024), 256, 0, stream>>>(x3, 256, ffn_w1, 1024, hid, 1024,
                                                    T, 1024, 256, ffn_b1, nullptr, 1);
  gemm_wmma<<<gemm_grid(T, 256), 256, 0, stream>>>(hid, 1024, ffn_w2, 256, (float*)d_out, 256,
                                                   T, 256, 1024, ffn_b2, q3, 0);
}
